// EnhancedDDNet_ODE_63660005262150
// MI455X (gfx1250) — matrix operand order repaired and checked by reading, not compiled, not run on hardware
//
#include <hip/hip_runtime.h>

// ---------------------------------------------------------------------------
// EnhancedDDNet-ODE forward, fused for CDNA5 (gfx1250, wave32, WMMA).
// Dominant cost: 3 branches x 63 RK4 steps x 8 GEMMs of (8192x128)@(128x128).
// Strategy: one persistent kernel per branch keeps y, RK4 accumulator and both
// f16 weight matrices in LDS and runs the whole scan with
// v_wmma_f32_16x16x32_f16 (f32 accumulate). Remaining ~3% of FLOPs use a
// generic WMMA GEMM (fuse/head projections) + small VALU kernels.
// Sigmoid/silu use v_exp_f32 + v_rcp_f32 (hardware transcendentals) instead of
// IEEE-precise division chains.
// ---------------------------------------------------------------------------

typedef __attribute__((ext_vector_type(16))) _Float16 v16h;
typedef __attribute__((ext_vector_type(8)))  _Float16 v8h;
typedef __attribute__((ext_vector_type(8)))  float    v8f;

static constexpr int kB = 128;          // batch
static constexpr int kT = 64;           // time steps
static constexpr int kF = 128;          // features
static constexpr int kM = kB * kT;      // 8192 rows
static constexpr int kNCLS = 60;
static constexpr float kEPS = 1e-5f;

static constexpr int RK_ROWS = 32;      // rows per workgroup in fused RK4
static constexpr int GM_ROWS = 64;      // rows per workgroup in generic GEMM

// ---------------------------------------------------------------------------
// Device helpers
// ---------------------------------------------------------------------------
__device__ __forceinline__ float dev_sigmoid(float x) {
  // v_exp_f32 + v_rcp_f32: 1/(1+e^-x) without the precise-division fixup chain
  return __builtin_amdgcn_rcpf(1.0f + __expf(-x));
}
__device__ __forceinline__ float dev_silu(float x) { return x * dev_sigmoid(x); }

__device__ __forceinline__ v8f wmma16(v16h a, v16h b, v8f c) {
  // D = A(16x32,f16) * B(32x16,f16) + C(16x16,f32)
  return __builtin_amdgcn_wmma_f32_16x16x32_f16(
      /*neg_a=*/false, a, /*neg_b=*/false, b,
      /*c_mod=*/(short)0, c, /*reuse_a=*/false, /*reuse_b=*/false);
}

// A fragment: row-major f16 tile in LDS, ld = row stride in halves.
// A 16x32 layout (matches B): lanes 0-15 hold K 0..15, lanes 16-31 hold
// K 16..31 for row M = lane&15.
__device__ __forceinline__ v16h load_a_frag(const _Float16* A, int m0, int k0,
                                            int ld, int lane) {
  const int mr = lane & 15;
  const int c0 = (lane & 16) ? 16 : 0;
  const _Float16* p = A + (size_t)(m0 + mr) * ld + k0 + c0;
  v8h lo = *(const v8h*)(p);
  v8h hi = *(const v8h*)(p + 8);
  v16h a;
#pragma unroll
  for (int i = 0; i < 8; ++i) { a[i] = lo[i]; a[i + 8] = hi[i]; }
  return a;
}

// B fragment: weights stored TRANSPOSED in LDS as Wt[n][k] (ld = K in halves)
// so the 16 K-halves a lane needs are contiguous -> ds_load_b128 pairs.
// B 32x16 layout used here (k order matches A): lanes 0-15 hold K 0..15, lanes 16-31 hold K 16..31,
// for column N = lane&15.
__device__ __forceinline__ v16h load_b_frag(const _Float16* Wt, int n0, int k0,
                                            int ld, int lane) {
  const int nr = lane & 15;
  const int koff = (lane & 16) ? 16 : 0;
  const _Float16* p = Wt + (size_t)(n0 + nr) * ld + k0 + koff;
  return *(const v16h*)p;
}

// C/D mapping: element v of v8f -> row m0 + v + ((lane&16)?8:0), col n0+(lane&15)

// ---------------------------------------------------------------------------
// Fused RK4 ODE branch: y evolves through 63 RK4 steps entirely in LDS.
//   _ode_f: z = LN(silu(y@W1+b1)@W2+b2)*g+b ; k = z*sigmoid(tw*t+tb)
// Each WG owns RK_ROWS rows of the (8192 x 128) state.
// ---------------------------------------------------------------------------
__global__ __launch_bounds__(256) void rk4_fused_kernel(
    const float* __restrict__ xin, float* __restrict__ yout,
    const float* __restrict__ W1, const float* __restrict__ b1,
    const float* __restrict__ W2, const float* __restrict__ b2,
    const float* __restrict__ lng, const float* __restrict__ lnb,
    const float* __restrict__ twp, const float* __restrict__ tbp) {
  extern __shared__ char smem[];
  _Float16* w1t = (_Float16*)smem;                 // [128][128] transposed
  _Float16* w2t = w1t + kF * kF;                   // [128][128] transposed
  float* ybuf = (float*)(w2t + kF * kF);           // RK_ROWS*128 carry y (f32)
  float* accb = ybuf + RK_ROWS * kF;               // RK4 k-accumulator
  float* zbuf = accb + RK_ROWS * kF;               // GEMM2 output staging
  _Float16* abuf = (_Float16*)(zbuf + RK_ROWS * kF); // GEMM1 A operand (f16)
  _Float16* hbuf = abuf + RK_ROWS * kF;            // GEMM2 A operand (f16)
  float* pr1 = (float*)(hbuf + RK_ROWS * kF);      // LN partial sums
  float* pr2 = pr1 + RK_ROWS * 8;
  float* b1c = pr2 + RK_ROWS * 8;                  // cached per-feature params
  float* b2c = b1c + kF;
  float* lgc = b2c + kF;
  float* lbc = lgc + kF;
  float* twc = lbc + kF;
  float* tbc = twc + kF;

  const int tid = threadIdx.x;
  const int row0 = blockIdx.x * RK_ROWS;

  // Stage weights transposed (f32 -> f16) and per-feature params into LDS.
  for (int e = tid; e < kF * kF; e += 256) {
    const int n = e >> 7, k = e & 127;           // dest wt[n*128 + k]
    w1t[e] = (_Float16)W1[k * kF + n];
    w2t[e] = (_Float16)W2[k * kF + n];
  }
  if (tid < kF) {
    b1c[tid] = b1[tid]; b2c[tid] = b2[tid];
    lgc[tid] = lng[tid]; lbc[tid] = lnb[tid];
    twc[tid] = twp[tid]; tbc[tid] = tbp[tid];
  }
  // Stage initial state.
  for (int e = tid; e < RK_ROWS * kF; e += 256) {
    const float v = xin[(size_t)row0 * kF + e];
    ybuf[e] = v;
    abuf[e] = (_Float16)v;
  }
  __syncthreads();

  const int wave = tid >> 5;
  const int lane = tid & 31;
  const int n0 = wave * 16;                       // 8 waves cover 128 cols
  const int col = n0 + (lane & 15);
  const int rsub = (lane & 16) ? 8 : 0;

  const float hstep = 1.0f / 63.0f;
  for (int s = 0; s < 63; ++s) {
    const float t0 = s * hstep;
    for (int ev = 0; ev < 4; ++ev) {
      const float t = t0 + ((ev == 0) ? 0.0f : (ev == 3) ? hstep : 0.5f * hstep);

      // ---- GEMM1: h = silu(ystage @ W1 + b1) -> hbuf (f16) ----
      {
        const float bn = b1c[col];
#pragma unroll
        for (int mt = 0; mt < RK_ROWS / 16; ++mt) {
          v8f c;
#pragma unroll
          for (int v = 0; v < 8; ++v) c[v] = bn;
#pragma unroll
          for (int kt = 0; kt < 4; ++kt) {
            v16h a = load_a_frag(abuf, mt * 16, kt * 32, kF, lane);
            v16h b = load_b_frag(w1t, n0, kt * 32, kF, lane);
            c = wmma16(a, b, c);
          }
          const int rb = mt * 16 + rsub;
#pragma unroll
          for (int v = 0; v < 8; ++v)
            hbuf[(rb + v) * kF + col] = (_Float16)dev_silu(c[v]);
        }
      }
      __syncthreads();

      // ---- GEMM2: z = h @ W2 + b2 -> zbuf (f32) ----
      {
        const float bn = b2c[col];
#pragma unroll
        for (int mt = 0; mt < RK_ROWS / 16; ++mt) {
          v8f c;
#pragma unroll
          for (int v = 0; v < 8; ++v) c[v] = bn;
#pragma unroll
          for (int kt = 0; kt < 4; ++kt) {
            v16h a = load_a_frag(hbuf, mt * 16, kt * 32, kF, lane);
            v16h b = load_b_frag(w2t, n0, kt * 32, kF, lane);
            c = wmma16(a, b, c);
          }
          const int rb = mt * 16 + rsub;
#pragma unroll
          for (int v = 0; v < 8; ++v)
            zbuf[(rb + v) * kF + col] = c[v];
        }
      }
      __syncthreads();

      // ---- per-row LayerNorm + time gate + RK4 combine ----
      {
        const int row = tid >> 3;      // 32 rows x 8 threads
        const int seg = tid & 7;       // 16 columns each
        const float* zr = zbuf + row * kF;
        float sm = 0.0f, ss = 0.0f;
#pragma unroll
        for (int i = 0; i < 16; ++i) {
          const float v = zr[seg * 16 + i];
          sm += v; ss += v * v;
        }
        pr1[tid] = sm; pr2[tid] = ss;
        __syncthreads();
        float mu = 0.0f, m2 = 0.0f;
#pragma unroll
        for (int j = 0; j < 8; ++j) { mu += pr1[(row << 3) + j]; m2 += pr2[(row << 3) + j]; }
        mu *= (1.0f / kF);
        const float var = m2 * (1.0f / kF) - mu * mu;
        const float rinv = rsqrtf(var + kEPS);
#pragma unroll
        for (int i = 0; i < 16; ++i) {
          const int c = seg * 16 + i;
          const int ix = row * kF + c;
          const float zn = (zr[c] - mu) * rinv * lgc[c] + lbc[c];
          const float kk = zn * dev_sigmoid(twc[c] * t + tbc[c]);
          const float yv = ybuf[ix];
          float stage;
          if (ev == 0)      { accb[ix] = kk;                 stage = yv + 0.5f * hstep * kk; }
          else if (ev == 1) { accb[ix] += 2.0f * kk;         stage = yv + 0.5f * hstep * kk; }
          else if (ev == 2) { accb[ix] += 2.0f * kk;         stage = yv + hstep * kk; }
          else {
            const float acc = accb[ix] + kk;
            const float yn = yv + (hstep / 6.0f) * acc;
            ybuf[ix] = yn;  stage = yn;
          }
          abuf[ix] = (_Float16)stage;
        }
      }
      __syncthreads();
    }
  }

  for (int e = tid; e < RK_ROWS * kF; e += 256)
    yout[(size_t)row0 * kF + e] = ybuf[e];
}

// ---------------------------------------------------------------------------
// Generic WMMA GEMM, N = 128 fixed: C = act(A(MxK) @ W(Kx128) + bias)
// W transposed into LDS once per WG; A staged in K-chunks of 32.
// ---------------------------------------------------------------------------
__global__ __launch_bounds__(256) void gemm_wmma_n128_kernel(
    const float* __restrict__ A, const float* __restrict__ W,
    const float* __restrict__ bias, float* __restrict__ C, int M, int K, int act) {
  extern __shared__ char smem[];
  _Float16* wt = (_Float16*)smem;                 // [128][K] transposed
  _Float16* ab = wt + (size_t)kF * K;             // [GM_ROWS][32]
  float* bc = (float*)(ab + GM_ROWS * 32);        // [128]

  const int tid = threadIdx.x;
  const int row0 = blockIdx.x * GM_ROWS;

  for (int e = tid; e < kF * K; e += 256) {
    const int n = e / K, k = e - n * K;
    wt[e] = (_Float16)W[(size_t)k * kF + n];
  }
  if (tid < kF) bc[tid] = bias[tid];
  __syncthreads();

  const int wave = tid >> 5, lane = tid & 31;
  const int n0 = wave * 16;
  const int col = n0 + (lane & 15);
  const int rsub = (lane & 16) ? 8 : 0;

  v8f acc[GM_ROWS / 16];
  {
    const float bn = bc[col];
#pragma unroll
    for (int mt = 0; mt < GM_ROWS / 16; ++mt)
#pragma unroll
      for (int v = 0; v < 8; ++v) acc[mt][v] = bn;
  }

  const int nkt = K / 32;
  for (int kt = 0; kt < nkt; ++kt) {
    for (int e = tid; e < GM_ROWS * 32; e += 256) {
      const int r = e >> 5, c = e & 31;
      ab[e] = (_Float16)A[(size_t)(row0 + r) * K + kt * 32 + c];
      if (kt + 1 < nkt)
        __builtin_prefetch(&A[(size_t)(row0 + r) * K + (kt + 1) * 32 + c], 0, 0);
    }
    __syncthreads();
#pragma unroll
    for (int mt = 0; mt < GM_ROWS / 16; ++mt) {
      v16h a = load_a_frag(ab, mt * 16, 0, 32, lane);
      v16h b = load_b_frag(wt, n0, kt * 32, K, lane);
      acc[mt] = wmma16(a, b, acc[mt]);
    }
    __syncthreads();
  }

#pragma unroll
  for (int mt = 0; mt < GM_ROWS / 16; ++mt) {
    const int rb = row0 + mt * 16 + rsub;
#pragma unroll
    for (int v = 0; v < 8; ++v) {
      float o = acc[mt][v];
      if (act == 1) o = dev_silu(o);
      C[(size_t)(rb + v) * kF + col] = o;
    }
  }
}

// ---------------------------------------------------------------------------
// Small VALU kernels (≈3% of FLOPs)
// ---------------------------------------------------------------------------
__global__ void conv1d_kernel(const float* __restrict__ x,   // (B,T,F)
                              const float* __restrict__ w,   // (F,F,k)
                              const float* __restrict__ bias,
                              float* __restrict__ y, int ktap) {
  const int idx = blockIdx.x * blockDim.x + threadIdx.x;
  if (idx >= kB * kT * kF) return;
  const int co = idx & 127, t = (idx >> 7) & 63, b = idx >> 13;
  const int pad = ktap / 2;
  float acc = bias[co];
  for (int tau = 0; tau < ktap; ++tau) {
    const int tt = t + tau - pad;
    if (tt < 0 || tt >= kT) continue;
    const float* xr = x + (size_t)(b * kT + tt) * kF;
    const float* wr = w + (size_t)co * kF * ktap + tau;
    for (int ci = 0; ci < kF; ++ci) acc += wr[ci * ktap] * xr[ci];
  }
  y[idx] = acc;
}

__global__ void colstats_kernel(const float* __restrict__ X, int M, int ldx,
                                float* __restrict__ mu, float* __restrict__ var) {
  __shared__ float s1[256], s2[256];
  const int j = blockIdx.x, tid = threadIdx.x;
  float a = 0.0f, b = 0.0f;
  for (int r = tid; r < M; r += 256) {
    const float v = X[(size_t)r * ldx + j];
    a += v; b += v * v;
  }
  s1[tid] = a; s2[tid] = b; __syncthreads();
  for (int s = 128; s > 0; s >>= 1) {
    if (tid < s) { s1[tid] += s1[tid + s]; s2[tid] += s2[tid + s]; }
    __syncthreads();
  }
  if (tid == 0) {
    const float m = s1[0] / M;
    mu[j] = m; var[j] = s2[0] / M - m * m;
  }
}

__global__ void norm_apply_kernel(const float* __restrict__ X, float* __restrict__ Y,
                                  int M, int N, int ldx, int ldy,
                                  const float* __restrict__ mu, const float* __restrict__ var,
                                  const float* __restrict__ g, const float* __restrict__ beta,
                                  int act) {
  const int idx = blockIdx.x * blockDim.x + threadIdx.x;
  if (idx >= M * N) return;
  const int r = idx / N, c = idx - r * N;
  float v = (X[(size_t)r * ldx + c] - mu[c]) * rsqrtf(var[c] + kEPS) * g[c] + beta[c];
  if (act) v = dev_silu(v);
  Y[(size_t)r * ldy + c] = v;
}

__global__ void dense_kernel(const float* __restrict__ A, const float* __restrict__ W,
                             const float* __restrict__ bias, float* __restrict__ C,
                             int M, int K, int N, int act) {
  const int idx = blockIdx.x * blockDim.x + threadIdx.x;
  if (idx >= M * N) return;
  const int r = idx / N, c = idx - r * N;
  float acc = bias ? bias[c] : 0.0f;
  const float* ar = A + (size_t)r * K;
  for (int k = 0; k < K; ++k) acc += ar[k] * W[(size_t)k * N + c];
  if (act == 1) acc = fmaxf(acc, 0.0f);
  C[idx] = acc;
}

// out[r, c (ld=ldo)] = x[r,c] * sigmoid(u[r,:] @ W + b[c])
__global__ void dense_gate_mul_kernel(const float* __restrict__ x, const float* __restrict__ u,
                                      const float* __restrict__ W, const float* __restrict__ b,
                                      float* __restrict__ out, int M, int K, int N, int ldo) {
  const int idx = blockIdx.x * blockDim.x + threadIdx.x;
  if (idx >= M * N) return;
  const int r = idx / N, c = idx - r * N;
  float acc = b[c];
  const float* ur = u + (size_t)r * K;
  for (int k = 0; k < K; ++k) acc += ur[k] * W[(size_t)k * N + c];
  out[(size_t)r * ldo + c] = x[(size_t)r * N + c] * dev_sigmoid(acc);
}

__global__ void poses_motion_kernel(const float* __restrict__ P,  // (B,T,J,D)=B,T,128
                                    float* __restrict__ slow, float* __restrict__ fast) {
  const int idx = blockIdx.x * blockDim.x + threadIdx.x;
  if (idx >= kB * kT * kF) return;
  const int jd = idx & 127, t = (idx >> 7) & 63, b = idx >> 13;
  const float* Pb = P + (size_t)b * kT * kF;
  const int q = (t * (kT - 1)) / kT;                       // slow index 0..62
  slow[idx] = Pb[(q + 1) * kF + jd] - Pb[q * kF + jd];
  const int tf = kT / 2;
  const int qf = ((t / 2) * (tf - 1)) / tf;                // fast index 0..30
  fast[idx] = Pb[(2 * (qf + 1)) * kF + jd] - Pb[(2 * qf) * kF + jd];
}

__global__ void maxpool_t_kernel(const float* __restrict__ X, float* __restrict__ Y) {
  const int idx = blockIdx.x * blockDim.x + threadIdx.x;
  if (idx >= kB * kF) return;
  const int c = idx & 127, b = idx >> 7;
  float m = -3.0e38f;
  for (int t = 0; t < kT; ++t) m = fmaxf(m, X[(size_t)(b * kT + t) * kF + c]);
  Y[idx] = m;
}

// ---------------------------------------------------------------------------
// Host-side orchestration
// ---------------------------------------------------------------------------
struct BranchP {
  const float *sW1, *sW2, *sb1, *sb2, *tW1, *tW2, *tb1, *tb2;   // attn
  const float *cb[3], *cbeta[3], *cg[3], *cw[3];                // conv (k=3,5,7)
  const float *fuW, *fub, *fubeta, *fug;                        // fuse
  const float *W1, *W2, *b1, *b2, *ln_b, *ln_g, *tb, *tw;       // ode
};
struct HeadP {
  const float *cW1, *cW2, *cb1, *cb2, *cbeta, *cg, *fW, *fb, *fbeta, *fg;
};

// jax tree-flatten order (dict keys sorted): branch = attn(8), conv(3x4:
// b,beta,g,w), fuse(W,b,beta,g), ode(W1,W2,b1,b2,ln_b,ln_g,tb,tw)
static BranchP load_branch(void* const* d, int i) {
  BranchP p;
  p.sW1 = (const float*)d[i + 0]; p.sW2 = (const float*)d[i + 1];
  p.sb1 = (const float*)d[i + 2]; p.sb2 = (const float*)d[i + 3];
  p.tW1 = (const float*)d[i + 4]; p.tW2 = (const float*)d[i + 5];
  p.tb1 = (const float*)d[i + 6]; p.tb2 = (const float*)d[i + 7];
  for (int c = 0; c < 3; ++c) {
    p.cb[c]    = (const float*)d[i + 8 + 4 * c + 0];
    p.cbeta[c] = (const float*)d[i + 8 + 4 * c + 1];
    p.cg[c]    = (const float*)d[i + 8 + 4 * c + 2];
    p.cw[c]    = (const float*)d[i + 8 + 4 * c + 3];
  }
  p.fuW = (const float*)d[i + 20]; p.fub = (const float*)d[i + 21];
  p.fubeta = (const float*)d[i + 22]; p.fug = (const float*)d[i + 23];
  p.W1 = (const float*)d[i + 24]; p.W2 = (const float*)d[i + 25];
  p.b1 = (const float*)d[i + 26]; p.b2 = (const float*)d[i + 27];
  p.ln_b = (const float*)d[i + 28]; p.ln_g = (const float*)d[i + 29];
  p.tb = (const float*)d[i + 30]; p.tw = (const float*)d[i + 31];
  return p;
}
static HeadP load_head(void* const* d, int i) {
  HeadP h;
  h.cW1 = (const float*)d[i + 0]; h.cW2 = (const float*)d[i + 1];
  h.cb1 = (const float*)d[i + 2]; h.cb2 = (const float*)d[i + 3];
  h.cbeta = (const float*)d[i + 4]; h.cg = (const float*)d[i + 5];
  h.fW = (const float*)d[i + 6]; h.fb = (const float*)d[i + 7];
  h.fbeta = (const float*)d[i + 8]; h.fg = (const float*)d[i + 9];
  return h;
}

static constexpr size_t RK4_LDS =
    2 * kF * kF * sizeof(_Float16)      // w1t, w2t
  + 3 * RK_ROWS * kF * sizeof(float)    // ybuf, accb, zbuf
  + 2 * RK_ROWS * kF * sizeof(_Float16) // abuf, hbuf
  + 2 * RK_ROWS * 8 * sizeof(float)     // pr1, pr2
  + 6 * kF * sizeof(float);             // cached params

static inline size_t gemm_lds(int K) {
  return (size_t)kF * K * sizeof(_Float16) + (size_t)GM_ROWS * 32 * sizeof(_Float16)
       + kF * sizeof(float);
}

struct Ws {
  float *slowbuf, *fastbuf, *ybuf, *convraw, *convcat, *fusebuf, *ubuf;
  float *headcat, *fwbuf, *pooled, *c1buf, *mu, *var;
};

static void run_branch(const BranchP& p, const float* xin, float* headcol,
                       const Ws& w, hipStream_t s) {
  const int ew = 256;
  // 1) fused RK4 scan (WMMA)
  rk4_fused_kernel<<<dim3(kM / RK_ROWS), dim3(256), RK4_LDS, s>>>(
      xin, w.ybuf, p.W1, p.b1, p.W2, p.b2, p.ln_g, p.ln_b, p.tw, p.tb);
  // 2) conv tower -> batch-stat norm -> silu -> concat (B,T,3F)
  const int ktaps[3] = {3, 5, 7};
  for (int c = 0; c < 3; ++c) {
    conv1d_kernel<<<dim3((kM * kF + ew - 1) / ew), dim3(ew), 0, s>>>(
        w.ybuf, p.cw[c], p.cb[c], w.convraw, ktaps[c]);
    colstats_kernel<<<dim3(kF), dim3(256), 0, s>>>(w.convraw, kM, kF, w.mu, w.var);
    norm_apply_kernel<<<dim3((kM * kF + ew - 1) / ew), dim3(ew), 0, s>>>(
        w.convraw, w.convcat + c * kF, kM, kF, kF, 3 * kF,
        w.mu, w.var, p.cg[c], p.cbeta[c], 1);
  }
  // 3) fuse projection (WMMA GEMM, K=384) -> batch-stat norm -> silu
  gemm_wmma_n128_kernel<<<dim3(kM / GM_ROWS), dim3(256), gemm_lds(3 * kF), s>>>(
      w.convcat, p.fuW, p.fub, w.fusebuf, kM, 3 * kF, 0);
  colstats_kernel<<<dim3(kF), dim3(256), 0, s>>>(w.fusebuf, kM, kF, w.mu, w.var);
  norm_apply_kernel<<<dim3((kM * kF + ew - 1) / ew), dim3(ew), 0, s>>>(
      w.fusebuf, w.fusebuf, kM, kF, kF, kF, w.mu, w.var, p.fug, p.fubeta, 1);
  // 4) two squeeze-excite gates; second writes into the head concat slab
  dense_kernel<<<dim3((kM * 16 + ew - 1) / ew), dim3(ew), 0, s>>>(
      w.fusebuf, p.sW1, p.sb1, w.ubuf, kM, kF, 16, 1);
  dense_gate_mul_kernel<<<dim3((kM * kF + ew - 1) / ew), dim3(ew), 0, s>>>(
      w.fusebuf, w.ubuf, p.sW2, p.sb2, w.fusebuf, kM, 16, kF, kF);
  dense_kernel<<<dim3((kM * 16 + ew - 1) / ew), dim3(ew), 0, s>>>(
      w.fusebuf, p.tW1, p.tb1, w.ubuf, kM, kF, 16, 1);
  dense_gate_mul_kernel<<<dim3((kM * kF + ew - 1) / ew), dim3(ew), 0, s>>>(
      w.fusebuf, w.ubuf, p.tW2, p.tb2, headcol, kM, 16, kF, 3 * kF);
}

extern "C" void kernel_launch(void* const* d_in, const int* in_sizes, int n_in,
                              void* d_out, int out_size, void* d_ws, size_t ws_size,
                              hipStream_t stream) {
  if (n_in < 108) return;
  const float* M = (const float*)d_in[0];
  const float* P = (const float*)d_in[1];
  const BranchP fast = load_branch(d_in, 2);
  const HeadP   head = load_head(d_in, 34);
  const BranchP jcd  = load_branch(d_in, 44);
  const BranchP slow = load_branch(d_in, 76);

  // carve workspace
  float* base = (float*)d_ws;
  size_t off = 0;
  auto take = [&](size_t n) { float* p = base + off; off += n; return p; };
  Ws w;
  w.slowbuf = take((size_t)kM * kF);
  w.fastbuf = take((size_t)kM * kF);
  w.ybuf    = take((size_t)kM * kF);
  w.convraw = take((size_t)kM * kF);
  w.convcat = take((size_t)kM * 3 * kF);
  w.fusebuf = take((size_t)kM * kF);
  w.ubuf    = take((size_t)kM * 16);
  w.headcat = take((size_t)kM * 3 * kF);
  w.fwbuf   = take((size_t)kM * kF);
  w.pooled  = take((size_t)kB * kF);
  w.c1buf   = take((size_t)kB * 64);
  w.mu      = take(512);
  w.var     = take(512);
  (void)ws_size; (void)in_sizes; (void)out_size;

  (void)hipFuncSetAttribute((const void*)rk4_fused_kernel,
                            hipFuncAttributeMaxDynamicSharedMemorySize, (int)RK4_LDS);
  (void)hipFuncSetAttribute((const void*)gemm_wmma_n128_kernel,
                            hipFuncAttributeMaxDynamicSharedMemorySize, (int)gemm_lds(3 * kF));

  const int ew = 256;
  poses_motion_kernel<<<dim3((kM * kF + ew - 1) / ew), dim3(ew), 0, stream>>>(
      P, w.slowbuf, w.fastbuf);

  run_branch(jcd,  M,         w.headcat + 0 * kF, w, stream);
  run_branch(slow, w.slowbuf, w.headcat + 1 * kF, w, stream);
  run_branch(fast, w.fastbuf, w.headcat + 2 * kF, w, stream);

  // head: 384 -> 128 projection (WMMA), batch norm + silu, max over T
  gemm_wmma_n128_kernel<<<dim3(kM / GM_ROWS), dim3(256), gemm_lds(3 * kF), stream>>>(
      w.headcat, head.fW, head.fb, w.fwbuf, kM, 3 * kF, 0);
  colstats_kernel<<<dim3(kF), dim3(256), 0, stream>>>(w.fwbuf, kM, kF, w.mu, w.var);
  norm_apply_kernel<<<dim3((kM * kF + ew - 1) / ew), dim3(ew), 0, stream>>>(
      w.fwbuf, w.fwbuf, kM, kF, kF, kF, w.mu, w.var, head.fg, head.fbeta, 1);
  maxpool_t_kernel<<<dim3((kB * kF + ew - 1) / ew), dim3(ew), 0, stream>>>(
      w.fwbuf, w.pooled);
  // classifier
  dense_kernel<<<dim3((kB * 64 + ew - 1) / ew), dim3(ew), 0, stream>>>(
      w.pooled, head.cW1, head.cb1, w.c1buf, kB, kF, 64, 0);
  colstats_kernel<<<dim3(64), dim3(256), 0, stream>>>(w.c1buf, kB, 64, w.mu, w.var);
  norm_apply_kernel<<<dim3((kB * 64 + ew - 1) / ew), dim3(ew), 0, stream>>>(
      w.c1buf, w.c1buf, kB, 64, 64, 64, w.mu, w.var, head.cg, head.cbeta, 1);
  dense_kernel<<<dim3((kB * kNCLS + ew - 1) / ew), dim3(ew), 0, stream>>>(
      w.c1buf, head.cW2, head.cb2, (float*)d_out, kB, 64, kNCLS, 0);
}
